// GNO_7275674599690
// MI455X (gfx1250) — compile-verified, matrix operand order checked by reading, not run on hardware
//
#include <hip/hip_runtime.h>
#include <hip/hip_bf16.h>

#define N_NODES 50000
#define N_EDGES 800000
#define HID 64
#define NLAYERS 4

typedef __attribute__((ext_vector_type(16))) __bf16 v16bf;
typedef __attribute__((ext_vector_type(8)))  __bf16 v8bf;
typedef __attribute__((ext_vector_type(8)))  float  v8f;

// ---------------- workspace layout (bytes, 256-aligned) ----------------
// hA  : N*64 bf16          @ 0          (6,400,000)
// hB  : N*64 bf16          @ 6,400,000
// agg : N*64 f32           @ 12,800,000 (12,800,000)
// we1 : L*4*6*512 bf16     @ 25,600,000 (98,304)   edge w1 fragments
// we2 : L*4*2*512 bf16     @ 25,698,304 (32,768)   edge w2 fragments
// wn1 : L*4*4*512 bf16     @ 25,731,072 (65,536)   node w1 fragments
// wn2 : L*4*2*512 bf16     @ 25,796,608 (32,768)   node w2 fragments
#define OFF_HA  0
#define OFF_HB  6400000
#define OFF_AGG 12800000
#define OFF_WE1 25600000
#define OFF_WE2 25698304
#define OFF_WN1 25731072
#define OFF_WN2 25796608

// Load one 16-element bf16 A-fragment from LDS: two contiguous 16B chunks.
static __device__ inline v16bf ld_frag(const __bf16* p0, const __bf16* p1) {
    v8bf lo = *(const v8bf*)p0;
    v8bf hi = *(const v8bf*)p1;
    v16bf r;
#pragma unroll
    for (int i = 0; i < 8; ++i) { r[i] = lo[i]; r[i + 8] = hi[i]; }
    return r;
}

// ---------------- weight pre-shuffle into B-fragment layout ----------------
// out[((l*4+nt)*KC + kc)*512 + lane*16 + j] = w[l][K][n]
//   n = nt*16 + (lane&15); koff = (lane<16)?0:8; K = kc*32 + koff + (j<8? j : j+8)
__global__ void prep_wfrag_kernel(const float* __restrict__ w, __bf16* __restrict__ out,
                                  int Ktot, int total) {
    int tid = blockIdx.x * 256 + threadIdx.x;
    if (tid >= total) return;
    const int KC = Ktot >> 5;
    const int per_l = 4 * KC * 512;
    const int l  = tid / per_l;
    int r        = tid % per_l;
    const int nt = r / (KC * 512);  r %= (KC * 512);
    const int kc = r / 512;         r %= 512;
    const int lane = r >> 4;
    const int j    = r & 15;
    const int koff = (lane < 16) ? 0 : 8;
    const int n    = nt * 16 + (lane & 15);
    const int K    = kc * 32 + koff + ((j < 8) ? j : j + 8);
    out[tid] = (__bf16)w[((size_t)l * Ktot + K) * 64 + n];
}

__global__ void zero_kernel(float* __restrict__ p, int total) {
    int i = blockIdx.x * 256 + threadIdx.x;
    if (i < total) p[i] = 0.0f;
}

// ---------------- encoder: h = feat @ enc_w + enc_b  (bf16 out) ----------------
__global__ void encode_kernel(const float* __restrict__ feat, const float* __restrict__ w,
                              const float* __restrict__ b, __bf16* __restrict__ h) {
    int tid = blockIdx.x * 256 + threadIdx.x;
    if (tid >= N_NODES * 32) return;
    const int n = tid >> 5, cp = tid & 31, c0 = cp * 2, c1 = c0 + 1;
    const float f0 = feat[n * 3 + 0], f1 = feat[n * 3 + 1], f2 = feat[n * 3 + 2];
    const float v0 = f0 * w[c0] + f1 * w[64 + c0] + f2 * w[128 + c0] + b[c0];
    const float v1 = f0 * w[c1] + f1 * w[64 + c1] + f2 * w[128 + c1] + b[c1];
    h[(size_t)n * 64 + c0] = (__bf16)v0;
    h[(size_t)n * 64 + c1] = (__bf16)v1;
}

// ---------------- edge MLP + scatter-add (per-wave 16-edge tile) ----------------
__global__ __launch_bounds__(128) void edge_msg_kernel(
    const float* __restrict__ pos, const long long* __restrict__ eidx,
    const float* __restrict__ eew, const float* __restrict__ eeb,
    const __bf16* __restrict__ h,
    const __bf16* __restrict__ w1f, const float* __restrict__ b1,
    const __bf16* __restrict__ w2f, const float* __restrict__ b2,
    float* __restrict__ agg)
{
    __shared__ __bf16 s_ain[4][16 * 192];   // [h[src] | h[dst] | e]
    __shared__ __bf16 s_rlu[4][16 * 64];
    __shared__ int    s_dst[4][16];
    const int wave = threadIdx.x >> 5;
    const int lane = threadIdx.x & 31;
    const int tile = blockIdx.x * 4 + wave;
    __bf16* ain = s_ain[wave];
    __bf16* rlu = s_rlu[wave];
    int*    dsts = s_dst[wave];

    const int m0   = lane & 15;
    const int koff = (lane >> 4) << 3;   // 0 or 8

    // -- fill 16x192 bf16 A tile (gathers + on-the-fly edge encoder) --
    for (int m = 0; m < 16; ++m) {
        const int ei = tile * 16 + m;
        const int s = (int)eidx[ei];
        const int d = (int)eidx[N_EDGES + ei];
        if (lane == 0) dsts[m] = d;
        ((unsigned int*)(ain + m * 192))[lane] =
            ((const unsigned int*)(h + (size_t)s * 64))[lane];
        ((unsigned int*)(ain + m * 192 + 64))[lane] =
            ((const unsigned int*)(h + (size_t)d * 64))[lane];
        const float p0 = pos[d * 3 + 0] - pos[s * 3 + 0];
        const float p1 = pos[d * 3 + 1] - pos[s * 3 + 1];
        const float p2 = pos[d * 3 + 2] - pos[s * 3 + 2];
        const int c0 = lane * 2, c1 = c0 + 1;
        const float e0 = p0 * eew[c0] + p1 * eew[64 + c0] + p2 * eew[128 + c0] + eeb[c0];
        const float e1 = p0 * eew[c1] + p1 * eew[64 + c1] + p2 * eew[128 + c1] + eeb[c1];
        ain[m * 192 + 128 + c0] = (__bf16)e0;
        ain[m * 192 + 128 + c1] = (__bf16)e1;
    }

    // -- GEMM1: [16x192] @ [192x64], bias, relu -> LDS tile --
    v16bf afr[6];
#pragma unroll
    for (int kc = 0; kc < 6; ++kc) {
        const __bf16* p = ain + m0 * 192 + kc * 32 + koff;
        afr[kc] = ld_frag(p, p + 16);
    }
#pragma unroll
    for (int nt = 0; nt < 4; ++nt) {
        v8f acc = {};
#pragma unroll
        for (int kc = 0; kc < 6; ++kc) {
            v16bf bf = *(const v16bf*)(w1f + (size_t)(nt * 6 + kc) * 512 + lane * 16);
            acc = __builtin_amdgcn_wmma_f32_16x16x32_bf16(
                false, afr[kc], false, bf, (short)0, acc, false, false);
        }
        const int col = nt * 16 + m0;
        const float bias = b1[col];
#pragma unroll
        for (int r = 0; r < 8; ++r) {
            float v = acc[r] + bias;
            v = v > 0.0f ? v : 0.0f;
            rlu[(r + koff) * 64 + col] = (__bf16)v;
        }
    }

    // -- GEMM2: [16x64] @ [64x64], bias, scatter-add into agg --
    v16bf a2[2];
#pragma unroll
    for (int kc = 0; kc < 2; ++kc) {
        const __bf16* p = rlu + m0 * 64 + kc * 32 + koff;
        a2[kc] = ld_frag(p, p + 16);
    }
#pragma unroll
    for (int nt = 0; nt < 4; ++nt) {
        v8f acc = {};
#pragma unroll
        for (int kc = 0; kc < 2; ++kc) {
            v16bf bf = *(const v16bf*)(w2f + (size_t)(nt * 2 + kc) * 512 + lane * 16);
            acc = __builtin_amdgcn_wmma_f32_16x16x32_bf16(
                false, a2[kc], false, bf, (short)0, acc, false, false);
        }
        const int col = nt * 16 + m0;
        const float bias = b2[col];
#pragma unroll
        for (int r = 0; r < 8; ++r) {
            const int d = dsts[r + koff];
            __hip_atomic_fetch_add(agg + (size_t)d * 64 + col, acc[r] + bias,
                                   __ATOMIC_RELAXED, __HIP_MEMORY_SCOPE_AGENT);
        }
    }
}

// ---------------- node MLP (per-wave 16-node tile) ----------------
__global__ __launch_bounds__(128) void node_upd_kernel(
    const __bf16* __restrict__ h, const float* __restrict__ agg,
    const __bf16* __restrict__ w1f, const float* __restrict__ b1,
    const __bf16* __restrict__ w2f, const float* __restrict__ b2,
    __bf16* __restrict__ hout, int ntiles)
{
    __shared__ __bf16 s_ain[4][16 * 128];   // [h | agg]
    __shared__ __bf16 s_out[4][16 * 64];    // relu tile, then output tile
    const int wave = threadIdx.x >> 5;
    const int lane = threadIdx.x & 31;
    const int tile = blockIdx.x * 4 + wave;
    if (tile >= ntiles) return;
    __bf16* ain = s_ain[wave];
    __bf16* tmp = s_out[wave];

    const int m0   = lane & 15;
    const int koff = (lane >> 4) << 3;

    // -- fill 16x128 bf16 tile --
    for (int m = 0; m < 16; ++m) {
        const int node = tile * 16 + m;
        ((unsigned int*)(ain + m * 128))[lane] =
            ((const unsigned int*)(h + (size_t)node * 64))[lane];
        const float a0 = agg[(size_t)node * 64 + lane * 2];
        const float a1 = agg[(size_t)node * 64 + lane * 2 + 1];
        ain[m * 128 + 64 + lane * 2]     = (__bf16)a0;
        ain[m * 128 + 64 + lane * 2 + 1] = (__bf16)a1;
    }

    // -- GEMM1: [16x128] @ [128x64], bias, relu -> LDS --
    v16bf afr[4];
#pragma unroll
    for (int kc = 0; kc < 4; ++kc) {
        const __bf16* p = ain + m0 * 128 + kc * 32 + koff;
        afr[kc] = ld_frag(p, p + 16);
    }
#pragma unroll
    for (int nt = 0; nt < 4; ++nt) {
        v8f acc = {};
#pragma unroll
        for (int kc = 0; kc < 4; ++kc) {
            v16bf bf = *(const v16bf*)(w1f + (size_t)(nt * 4 + kc) * 512 + lane * 16);
            acc = __builtin_amdgcn_wmma_f32_16x16x32_bf16(
                false, afr[kc], false, bf, (short)0, acc, false, false);
        }
        const int col = nt * 16 + m0;
        const float bias = b1[col];
#pragma unroll
        for (int r = 0; r < 8; ++r) {
            float v = acc[r] + bias;
            v = v > 0.0f ? v : 0.0f;
            tmp[(r + koff) * 64 + col] = (__bf16)v;
        }
    }

    // -- GEMM2: [16x64] @ [64x64], bias -> new h (via LDS for coalesced store) --
    v16bf a2[2];
#pragma unroll
    for (int kc = 0; kc < 2; ++kc) {
        const __bf16* p = tmp + m0 * 64 + kc * 32 + koff;
        a2[kc] = ld_frag(p, p + 16);
    }
#pragma unroll
    for (int nt = 0; nt < 4; ++nt) {
        v8f acc = {};
#pragma unroll
        for (int kc = 0; kc < 2; ++kc) {
            v16bf bf = *(const v16bf*)(w2f + (size_t)(nt * 2 + kc) * 512 + lane * 16);
            acc = __builtin_amdgcn_wmma_f32_16x16x32_bf16(
                false, a2[kc], false, bf, (short)0, acc, false, false);
        }
        const int col = nt * 16 + m0;
        const float bias = b2[col];
#pragma unroll
        for (int r = 0; r < 8; ++r)
            tmp[(r + koff) * 64 + col] = (__bf16)(acc[r] + bias);
    }
    for (int m = 0; m < 16; ++m) {
        const int node = tile * 16 + m;
        ((unsigned int*)(hout + (size_t)node * 64))[lane] =
            ((unsigned int*)(tmp + m * 64))[lane];
    }
}

// ---------------- decoder: out = h @ dec_w + dec_b ----------------
__global__ void decode_kernel(const __bf16* __restrict__ h, const float* __restrict__ w,
                              const float* __restrict__ b, float* __restrict__ out) {
    int n = blockIdx.x * 256 + threadIdx.x;
    if (n >= N_NODES) return;
    float acc = 0.0f;
#pragma unroll 8
    for (int j = 0; j < 64; ++j) acc += (float)h[(size_t)n * 64 + j] * w[j];
    out[n] = acc + b[0];
}

extern "C" void kernel_launch(void* const* d_in, const int* in_sizes, int n_in,
                              void* d_out, int out_size, void* d_ws, size_t ws_size,
                              hipStream_t stream) {
    const float*     pos  = (const float*)d_in[0];
    const float*     feat = (const float*)d_in[1];
    const long long* eidx = (const long long*)d_in[2];
    const float* enc_w = (const float*)d_in[3];
    const float* enc_b = (const float*)d_in[4];
    const float* eew   = (const float*)d_in[5];
    const float* eeb   = (const float*)d_in[6];
    const float* ew1   = (const float*)d_in[7];
    const float* eb1   = (const float*)d_in[8];
    const float* ew2   = (const float*)d_in[9];
    const float* eb2   = (const float*)d_in[10];
    const float* nw1   = (const float*)d_in[11];
    const float* nb1   = (const float*)d_in[12];
    const float* nw2   = (const float*)d_in[13];
    const float* nb2   = (const float*)d_in[14];
    const float* dec_w = (const float*)d_in[15];
    const float* dec_b = (const float*)d_in[16];

    char* ws = (char*)d_ws;
    __bf16* hA  = (__bf16*)(ws + OFF_HA);
    __bf16* hB  = (__bf16*)(ws + OFF_HB);
    float*  agg = (float*)(ws + OFF_AGG);
    __bf16* we1 = (__bf16*)(ws + OFF_WE1);
    __bf16* we2 = (__bf16*)(ws + OFF_WE2);
    __bf16* wn1 = (__bf16*)(ws + OFF_WN1);
    __bf16* wn2 = (__bf16*)(ws + OFF_WN2);

    // weight fragment prep (tiny)
    prep_wfrag_kernel<<<(NLAYERS*4*6*512 + 255) / 256, 256, 0, stream>>>(ew1, we1, 192, NLAYERS*4*6*512);
    prep_wfrag_kernel<<<(NLAYERS*4*2*512 + 255) / 256, 256, 0, stream>>>(ew2, we2,  64, NLAYERS*4*2*512);
    prep_wfrag_kernel<<<(NLAYERS*4*4*512 + 255) / 256, 256, 0, stream>>>(nw1, wn1, 128, NLAYERS*4*4*512);
    prep_wfrag_kernel<<<(NLAYERS*4*2*512 + 255) / 256, 256, 0, stream>>>(nw2, wn2,  64, NLAYERS*4*2*512);

    // encoder
    encode_kernel<<<(N_NODES * 32 + 255) / 256, 256, 0, stream>>>(feat, enc_w, enc_b, hA);

    __bf16* hin  = hA;
    __bf16* hout = hB;
    const int edge_tiles = N_EDGES / 16;        // 50000
    const int node_tiles = N_NODES / 16;        // 3125
    for (int l = 0; l < NLAYERS; ++l) {
        zero_kernel<<<(N_NODES * HID + 255) / 256, 256, 0, stream>>>(agg, N_NODES * HID);
        edge_msg_kernel<<<edge_tiles / 4, 128, 0, stream>>>(
            pos, eidx, eew, eeb, hin,
            we1 + (size_t)l * 4 * 6 * 512, eb1 + l * 64,
            we2 + (size_t)l * 4 * 2 * 512, eb2 + l * 64, agg);
        node_upd_kernel<<<(node_tiles + 3) / 4, 128, 0, stream>>>(
            hin, agg,
            wn1 + (size_t)l * 4 * 4 * 512, nb1 + l * 64,
            wn2 + (size_t)l * 4 * 2 * 512, nb2 + l * 64, hout, node_tiles);
        __bf16* t = hin; hin = hout; hout = t;
    }

    decode_kernel<<<(N_NODES + 255) / 256, 256, 0, stream>>>(hin, dec_w, dec_b, (float*)d_out);
}
